// MoE_46591805227314
// MI455X (gfx1250) — matrix operand order repaired and checked by reading, not compiled, not run on hardware
//
#include <hip/hip_runtime.h>
#include <hip/hip_bf16.h>
#include <math.h>

// ---------------- problem constants (from setup_inputs) ----------------
#define T_TOK   8192      // B*S = 4*2048
#define DMODEL  1024
#define NEXP    16
#define FFN     4096
#define CAP     512       // ceil(T/E * 1.0)

typedef __attribute__((ext_vector_type(16))) __bf16 v16bf;
typedef __attribute__((ext_vector_type(8)))  float  v8f;

union AFragU { v16bf v; uint4 u[2]; };
union PkBf   { __bf16 h[2]; unsigned u; };

#define LDS_BUF_ELEMS (128 * 32)   // one 128x32 bf16 tile

// ---------------- workspace layout (bytes) ----------------
#define ME_OFF   0u
#define CNT_OFF  4096u
#define IDX_OFF  8192u                       // int idx1[T]
#define GV_OFF   40960u                      // float gate_val[T]
#define TOS_OFF  73728u                      // int token_of_slot[E*CAP]
#define GOS_OFF  106496u                     // float gate_of_slot[E*CAP]
#define XE_OFF   139264u                     // __bf16 Xe[E*CAP*D]   (16 MB)
#define H_OFF    16916480u                   // __bf16 H [E*CAP*F]   (64 MB)

// ---------------- K0: init ----------------
__global__ __launch_bounds__(256) void k_init(float* out, int* tos, float* me, int* cnt) {
    int gid = blockIdx.x * 256 + threadIdx.x;
    int stride = gridDim.x * 256;
    for (int i = gid; i < T_TOK * DMODEL; i += stride) out[i] = 0.0f;
    if (gid < NEXP * CAP) tos[gid] = -1;
    if (gid < NEXP) { me[gid] = 0.0f; cnt[gid] = 0; }
}

// ---------------- K1: gating (logits -> softmax -> argmax) ----------------
__global__ __launch_bounds__(128) void k_gate(const float* __restrict__ x,
                                              const float* __restrict__ wg,
                                              int* __restrict__ idx1,
                                              float* __restrict__ gateval,
                                              float* __restrict__ me_g) {
    __shared__ float wgs[256 * NEXP];   // 16 KB chunk of wg
    __shared__ float bme[NEXP];
    const int tid = threadIdx.x;
    const int t = blockIdx.x * 128 + tid;
    if (tid < NEXP) bme[tid] = 0.0f;

    float acc[NEXP];
#pragma unroll
    for (int e = 0; e < NEXP; e++) acc[e] = 0.0f;

    const float* xrow = x + (size_t)t * DMODEL;
    for (int d0 = 0; d0 < DMODEL; d0 += 256) {
        __syncthreads();
        const float4* src4 = (const float4*)(wg + (size_t)d0 * NEXP);
        float4* dst4 = (float4*)wgs;
        for (int i = tid; i < 256 * NEXP / 4; i += 128) dst4[i] = src4[i];
        __syncthreads();
        for (int dd = 0; dd < 256; dd += 4) {
            float4 xv = *(const float4*)(xrow + d0 + dd);
            const float* wr = &wgs[dd * NEXP];
#pragma unroll
            for (int e = 0; e < NEXP; e++) {
                acc[e] += xv.x * wr[e] + xv.y * wr[NEXP + e] +
                          xv.z * wr[2 * NEXP + e] + xv.w * wr[3 * NEXP + e];
            }
        }
    }
    float m = acc[0]; int bi = 0;
#pragma unroll
    for (int e = 1; e < NEXP; e++) { if (acc[e] > m) { m = acc[e]; bi = e; } }
    float s = 0.0f;
#pragma unroll
    for (int e = 0; e < NEXP; e++) { acc[e] = __expf(acc[e] - m); s += acc[e]; }
    float inv = 1.0f / s;
    idx1[t] = bi;
    gateval[t] = inv;
    __syncthreads();
#pragma unroll
    for (int e = 0; e < NEXP; e++) atomicAdd(&bme[e], acc[e] * inv);
    __syncthreads();
    if (tid < NEXP) atomicAdd(&me_g[tid], bme[tid]);
}

// ---------------- K2: deterministic per-expert prefix scan + capacity drop ----------------
__global__ void k_scan(const int* __restrict__ idx1, const float* __restrict__ gateval,
                       int* __restrict__ tos, float* __restrict__ gos, int* __restrict__ cnt) {
    const int e = blockIdx.x;
    const int lane = threadIdx.x;   // 32 threads
    __shared__ int pred[32];
    int c = 0;
    for (int base = 0; base < T_TOK; base += 32) {
        int t = base + lane;
        int p = (idx1[t] == e) ? 1 : 0;
        pred[lane] = p;
        __syncthreads();
        int pre = 0, tot = 0;
#pragma unroll
        for (int j = 0; j < 32; j++) { int pj = pred[j]; tot += pj; if (j < lane) pre += pj; }
        if (p) {
            int pos = c + pre;
            if (pos < CAP) { tos[e * CAP + pos] = t; gos[e * CAP + pos] = gateval[t]; }
        }
        c += tot;
        __syncthreads();
    }
    if (lane == 0) cnt[e] = c;
}

// ---------------- K3: l_aux + exp_counts tail ----------------
__global__ void k_finalize(const float* __restrict__ me, const int* __restrict__ cnt,
                           float* __restrict__ out) {
    __shared__ float s[NEXP];
    int l = threadIdx.x;
    if (l < NEXP) {
        float mev = me[l] / (float)T_TOK;
        float cev = (float)cnt[l] / (float)T_TOK;
        s[l] = mev * cev;
        out[(size_t)T_TOK * DMODEL + 1 + l] = (float)cnt[l];
    }
    __syncthreads();
    if (l == 0) {
        float a = 0.0f;
        for (int e = 0; e < NEXP; e++) a += s[e];
        out[(size_t)T_TOK * DMODEL] = a * (float)NEXP;
    }
}

// ---------------- K4: dispatch tokens -> bf16 Xe[E,CAP,D] ----------------
__global__ __launch_bounds__(256) void k_dispatch(const float* __restrict__ x,
                                                  const int* __restrict__ tos,
                                                  __bf16* __restrict__ Xe) {
    const int slot = blockIdx.x;
    const int t = tos[slot];
    const int tid = threadIdx.x;
    __bf16* dst = Xe + (size_t)slot * DMODEL;
    union { __bf16 h[4]; uint2 u; } pk;
    if (t >= 0) {
        float4 v = *(const float4*)(x + (size_t)t * DMODEL + tid * 4);
        pk.h[0] = (__bf16)v.x; pk.h[1] = (__bf16)v.y; pk.h[2] = (__bf16)v.z; pk.h[3] = (__bf16)v.w;
    } else {
        pk.u = make_uint2(0u, 0u);
    }
    *(uint2*)&dst[tid * 4] = pk.u;
}

// branchless tanh-GELU:  gelu(v) = v * sigmoid(2 * 0.79788456*(v + 0.044715 v^3))
__device__ __forceinline__ float gelu_tanh(float v) {
    float u = 0.7978845608028654f * (v + 0.044715f * v * v * v);
    return __fdividef(v, 1.0f + __expf(-2.0f * u));
}

// ---------------- shared GEMM mainloop ----------------
// 128x128 block tile, BK=32, 8 waves, ping-pong LDS double buffer (1 barrier/K-step).
// Wave w: rows [(w&3)*32, +32) as 2 x 16-tiles; cols [(w>>2)*64, +64) as 4 x 16-tiles.
// A staging: thread -> (row am, 16 bf16 at ako).   B staging: thread -> (k-pair bkp, 8 cols at bn0),
// packed pairs go out as single ds_store_b32 via v_cvt_pk_bf16_f32.
template<int KDIM, int LDA, int LDB>
__device__ __forceinline__ void gemm_mainloop(const __bf16* __restrict__ Ag,
                                              const float* __restrict__ Bg,
                                              __bf16* As, __bf16* Bs,
                                              v8f acc[2][4]) {
    const int tid = threadIdx.x;
    const int wave = tid >> 5, lane = tid & 31;
    const int mw = (wave & 3) * 32, nw = (wave >> 2) * 64;
    const int mrow = lane & 15, hs = lane >> 4;

    const int am  = tid >> 1,       ako = (tid & 1) * 16;  // A: 128 rows x 32
    const int bn0 = (tid & 15) * 8, bkp = (tid >> 4) * 2;  // B: k-pair x 8 cols

    uint4  areg[2];
    float4 breg[4];   // breg[0..1] = row bkp cols bn0..bn0+7 ; breg[2..3] = row bkp+1

    auto load_regs = [&](int k0) {
        const __bf16* ap = Ag + (size_t)am * LDA + k0 + ako;
        areg[0] = *(const uint4*)ap;
        areg[1] = *(const uint4*)(ap + 8);
        const float* b0 = Bg + (size_t)(k0 + bkp) * LDB + bn0;
        const float* b1 = b0 + LDB;
        breg[0] = *(const float4*)b0;
        breg[1] = *(const float4*)(b0 + 4);
        breg[2] = *(const float4*)b1;
        breg[3] = *(const float4*)(b1 + 4);
        if (k0 + 32 < KDIM)
            __builtin_prefetch(Bg + (size_t)(k0 + 32 + bkp) * LDB + bn0, 0, 1);
    };

    auto store_stage = [&](int p) {
        __bf16* Asb = As + p * LDS_BUF_ELEMS;
        __bf16* Bsb = Bs + p * LDS_BUF_ELEMS;
        *(uint4*)&Asb[am * 32 + ako]     = areg[0];
        *(uint4*)&Asb[am * 32 + ako + 8] = areg[1];
        const float* r0 = (const float*)&breg[0];
        const float* r1 = (const float*)&breg[2];
#pragma unroll
        for (int c = 0; c < 8; c++) {
            PkBf pk;
            pk.h[0] = (__bf16)r0[c];
            pk.h[1] = (__bf16)r1[c];
            *(unsigned*)&Bsb[(bn0 + c) * 32 + bkp] = pk.u;
        }
    };

    auto compute = [&](int p) {
        const __bf16* Asb = As + p * LDS_BUF_ELEMS;
        const __bf16* Bsb = Bs + p * LDS_BUF_ELEMS;
        AFragU a0, a1;
        a0.u[0] = *(const uint4*)&Asb[(mw + mrow) * 32 + 8 * hs];
        a0.u[1] = *(const uint4*)&Asb[(mw + mrow) * 32 + 16 + 8 * hs];
        a1.u[0] = *(const uint4*)&Asb[(mw + 16 + mrow) * 32 + 8 * hs];
        a1.u[1] = *(const uint4*)&Asb[(mw + 16 + mrow) * 32 + 16 + 8 * hs];
#pragma unroll
        for (int j = 0; j < 4; j++) {
            AFragU b;
            int n = nw + j * 16 + mrow;
            b.u[0] = *(const uint4*)&Bsb[n * 32 + 8 * hs];
            b.u[1] = *(const uint4*)&Bsb[n * 32 + 16 + 8 * hs];
            acc[0][j] = __builtin_amdgcn_wmma_f32_16x16x32_bf16(
                false, a0.v, false, b.v, (short)0, acc[0][j], false, false);
            acc[1][j] = __builtin_amdgcn_wmma_f32_16x16x32_bf16(
                false, a1.v, false, b.v, (short)0, acc[1][j], false, false);
        }
    };

    // pipeline: regs(k0) -> LDS buf0 ; loop { prefetch regs(k0+32); compute buf p; stage buf 1-p; barrier }
    load_regs(0);
    store_stage(0);
    __syncthreads();
    for (int k0 = 0; k0 < KDIM; k0 += 32) {
        const int p = (k0 >> 5) & 1;
        const bool more = (k0 + 32) < KDIM;
        if (more) load_regs(k0 + 32);   // global loads overlap the 8 WMMAs below
        compute(p);
        if (more) store_stage(1 - p);
        __syncthreads();
    }
}

// ---------------- K5: GEMM1  H = gelu(Xe @ w1 + b1) ----------------
__global__ __launch_bounds__(256) void k_gemm1(const __bf16* __restrict__ Xe,
                                               const float* __restrict__ w1,
                                               const float* __restrict__ b1,
                                               __bf16* __restrict__ H) {
    const int e  = blockIdx.z;
    const int n0 = blockIdx.x * 128;
    const int m0 = blockIdx.y * 128;
    __shared__ __bf16 As[2 * LDS_BUF_ELEMS];
    __shared__ __bf16 Bs[2 * LDS_BUF_ELEMS];
    const int lane = threadIdx.x & 31, wave = threadIdx.x >> 5;
    const int mw = (wave & 3) * 32, nw = (wave >> 2) * 64;
    const int mrow = lane & 15, hs = lane >> 4;

    v8f acc[2][4];
#pragma unroll
    for (int i = 0; i < 2; i++)
#pragma unroll
        for (int j = 0; j < 4; j++) acc[i][j] = (v8f)(0.0f);

    gemm_mainloop<DMODEL, DMODEL, FFN>(
        Xe + ((size_t)e * CAP + m0) * DMODEL,
        w1 + (size_t)e * DMODEL * FFN + n0,
        As, Bs, acc);

#pragma unroll
    for (int j = 0; j < 4; j++) {
        int n = n0 + nw + j * 16 + mrow;
        float bb = b1[e * FFN + n];
#pragma unroll
        for (int i = 0; i < 2; i++) {
#pragma unroll
            for (int r = 0; r < 8; r++) {
                int m = m0 + mw + i * 16 + r + 8 * hs;
                float v = acc[i][j][r] + bb;
                H[((size_t)e * CAP + m) * FFN + n] = (__bf16)gelu_tanh(v);
            }
        }
    }
}

// ---------------- K6: GEMM2  out[t,:] = gate * (H @ w2 + b2)  fused combine ----------------
__global__ __launch_bounds__(256) void k_gemm2(const __bf16* __restrict__ H,
                                               const float* __restrict__ w2,
                                               const float* __restrict__ b2,
                                               const int* __restrict__ tos,
                                               const float* __restrict__ gos,
                                               float* __restrict__ out) {
    const int e  = blockIdx.z;
    const int n0 = blockIdx.x * 128;
    const int m0 = blockIdx.y * 128;
    __shared__ __bf16 As[2 * LDS_BUF_ELEMS];
    __shared__ __bf16 Bs[2 * LDS_BUF_ELEMS];
    const int lane = threadIdx.x & 31, wave = threadIdx.x >> 5;
    const int mw = (wave & 3) * 32, nw = (wave >> 2) * 64;
    const int mrow = lane & 15, hs = lane >> 4;

    v8f acc[2][4];
#pragma unroll
    for (int i = 0; i < 2; i++)
#pragma unroll
        for (int j = 0; j < 4; j++) acc[i][j] = (v8f)(0.0f);

    gemm_mainloop<FFN, FFN, DMODEL>(
        H + ((size_t)e * CAP + m0) * FFN,
        w2 + (size_t)e * FFN * DMODEL + n0,
        As, Bs, acc);

#pragma unroll
    for (int j = 0; j < 4; j++) {
        int n = n0 + nw + j * 16 + mrow;
        float bb = b2[e * DMODEL + n];
#pragma unroll
        for (int i = 0; i < 2; i++) {
#pragma unroll
            for (int r = 0; r < 8; r++) {
                int m = m0 + mw + i * 16 + r + 8 * hs;
                int slot = e * CAP + m;
                int t = tos[slot];
                if (t >= 0) out[(size_t)t * DMODEL + n] = gos[slot] * (acc[i][j][r] + bb);
            }
        }
    }
}

// ---------------- host launcher ----------------
extern "C" void kernel_launch(void* const* d_in, const int* in_sizes, int n_in,
                              void* d_out, int out_size, void* d_ws, size_t ws_size,
                              hipStream_t stream) {
    const float* x  = (const float*)d_in[0];
    const float* wg = (const float*)d_in[1];
    const float* w1 = (const float*)d_in[2];
    const float* b1 = (const float*)d_in[3];
    const float* w2 = (const float*)d_in[4];
    const float* b2 = (const float*)d_in[5];
    float* out = (float*)d_out;

    char* ws = (char*)d_ws;
    float* me    = (float*)(ws + ME_OFF);
    int*   cnt   = (int*)  (ws + CNT_OFF);
    int*   idx1  = (int*)  (ws + IDX_OFF);
    float* gv    = (float*)(ws + GV_OFF);
    int*   tos   = (int*)  (ws + TOS_OFF);
    float* gos   = (float*)(ws + GOS_OFF);
    __bf16* Xe   = (__bf16*)(ws + XE_OFF);
    __bf16* H    = (__bf16*)(ws + H_OFF);

    k_init<<<8192, 256, 0, stream>>>(out, tos, me, cnt);
    k_gate<<<T_TOK / 128, 128, 0, stream>>>(x, wg, idx1, gv, me);
    k_scan<<<NEXP, 32, 0, stream>>>(idx1, gv, tos, gos, cnt);
    k_finalize<<<1, 32, 0, stream>>>(me, cnt, out);
    k_dispatch<<<NEXP * CAP, 256, 0, stream>>>(x, tos, Xe);
    k_gemm1<<<dim3(FFN / 128, CAP / 128, NEXP), 256, 0, stream>>>(Xe, w1, b1, H);
    k_gemm2<<<dim3(DMODEL / 128, CAP / 128, NEXP), 256, 0, stream>>>(H, w2, b2, tos, gos, out);
}
